// DisMaxLossFirstPart_44770739093996
// MI455X (gfx1250) — matrix operand order repaired and checked by reading, not compiled, not run on hardware
//
#include <hip/hip_runtime.h>
#include <math.h>

// ---------------------------------------------------------------------------
// DisMax first part for MI455X (gfx1250), wave32 + WMMA bf16.
//   B=16384, F=1024, C=1000 (padded to 1024 for tiling)
// ws layout: [0)          bf16 normalized features   16384*1024
//            [16384*1024) bf16 normalized prototypes  1024*1024 (rows>=1000 zero)
// total ws: 35,651,584 bytes
// GEMM: block tile 64(M) x 256(N) x 32(K), 8 waves (2x4), 32x64 per wave
//       => 8 x v_wmma_f32_16x16x32_bf16 per wave per K-step.
//       Global->LDS staging via GLOBAL_LOAD_ASYNC_TO_LDS_B128 (ASYNCcnt).
// ---------------------------------------------------------------------------

typedef __attribute__((ext_vector_type(16))) __bf16 v16bf;
typedef __attribute__((ext_vector_type(8)))  float  v8f;
typedef __attribute__((ext_vector_type(4)))  int    v4i;

struct Q32 { uint4 a, b; };  // 32 bytes == one v16bf fragment

#if defined(__has_builtin)
#if __has_builtin(__builtin_amdgcn_global_load_async_to_lds_b128) && \
    __has_builtin(__builtin_amdgcn_s_wait_asynccnt)
#define USE_ASYNC_LDS 1
#endif
#endif

#ifndef USE_ASYNC_LDS
#define USE_ASYNC_LDS 0
#endif

#if USE_ASYNC_LDS
// global (AS1) -> LDS (AS3) 16-byte async copy, tracked by ASYNCcnt
__device__ __forceinline__ void async_copy16(const unsigned short* g, unsigned short* l) {
    __builtin_amdgcn_global_load_async_to_lds_b128(
        (__attribute__((address_space(1))) v4i*)(v4i*)(unsigned short*)g,
        (__attribute__((address_space(3))) v4i*)(v4i*)l, 0, 0);
}
#endif

__device__ __forceinline__ unsigned short f2bf(float f) {
    unsigned int u = __float_as_uint(f);
    u += 0x7FFFu + ((u >> 16) & 1u);          // round-to-nearest-even
    return (unsigned short)(u >> 16);
}

__device__ __forceinline__ float fast_sqrtf(float x) {
#if defined(__has_builtin)
#if __has_builtin(__builtin_amdgcn_sqrtf)
    return __builtin_amdgcn_sqrtf(x);         // raw v_sqrt_f32; x in [0,4]
#else
    return sqrtf(x);
#endif
#else
    return sqrtf(x);
#endif
}

// One block per row. Rows >= nkeep are zero-filled (prototype padding).
__global__ __launch_bounds__(256) void normalize_rows_kernel(
    const float* __restrict__ src, unsigned short* __restrict__ dst, int nkeep) {
    __shared__ float red[256];
    const int row = blockIdx.x;
    const int t   = threadIdx.x;
    unsigned short* drow = dst + (size_t)row * 1024;
    if (row >= nkeep) {
        *(uint2*)&drow[t * 4] = make_uint2(0u, 0u);
        return;
    }
    const float4 v = *(const float4*)(src + (size_t)row * 1024 + t * 4);
    red[t] = v.x * v.x + v.y * v.y + v.z * v.z + v.w * v.w;
    __syncthreads();
    for (int off = 128; off; off >>= 1) {
        if (t < off) red[t] += red[t + off];
        __syncthreads();
    }
    const float inv = 1.0f / fmaxf(sqrtf(red[0]), 1e-12f);
    unsigned int b0 = (unsigned int)f2bf(v.x * inv) | ((unsigned int)f2bf(v.y * inv) << 16);
    unsigned int b1 = (unsigned int)f2bf(v.z * inv) | ((unsigned int)f2bf(v.w * inv) << 16);
    *(uint2*)&drow[t * 4] = make_uint2(b0, b1);
}

__global__ __launch_bounds__(256) void gemm_iso_kernel(
    const unsigned short* __restrict__ wsA,   // [16384,1024] bf16 bits
    const unsigned short* __restrict__ wsP,   // [1024,1024]  bf16 bits
    const float* __restrict__ dscale,
    float* __restrict__ out) {
    // rows padded to 40 ushorts (80 B, multiple of 16 B) to keep chunks aligned
    __shared__ __align__(16) unsigned short sA[64 * 40];    //  5 KB
    __shared__ __align__(16) unsigned short sB[256 * 40];   // 20 KB

    const int tid  = threadIdx.x;
    const int wave = tid >> 5;
    const int lane = tid & 31;
    const int kh   = lane >> 4;     // lane half selects K group
    const int lr   = lane & 15;     // row (A) / col (B) within 16-tile
    const int wr   = wave >> 2;     // wave row  0..1  (32 rows each)
    const int wc   = wave & 3;      // wave col  0..3  (64 cols each)
    const int M0   = blockIdx.y * 64;
    const int N0   = blockIdx.x * 256;

    const unsigned short* gA = wsA + (size_t)M0 * 1024;
    const unsigned short* gB = wsP + (size_t)N0 * 1024;

    v8f acc[2][4];
#pragma unroll
    for (int i = 0; i < 2; ++i)
#pragma unroll
        for (int j = 0; j < 4; ++j) acc[i][j] = (v8f)0.0f;

    const int ar = tid >> 2;            // A copy: 256 chunks of 16 B
    const int ac = (tid & 3) * 8;

    for (int k0 = 0; k0 < 1024; k0 += 32) {
#if USE_ASYNC_LDS
        // CDNA5 async copy: global -> LDS without VGPR round-trip (ASYNCcnt).
        async_copy16(&gA[(size_t)ar * 1024 + k0 + ac], &sA[ar * 40 + ac]);
#pragma unroll
        for (int h = 0; h < 4; ++h) {   // B: 1024 chunks of 16 B
            const int c  = tid + h * 256;
            const int br = c >> 2;
            const int bc = (c & 3) * 8;
            async_copy16(&gB[(size_t)br * 1024 + k0 + bc], &sB[br * 40 + bc]);
        }
        __builtin_amdgcn_s_wait_asynccnt(0);
#else
        *(uint4*)&sA[ar * 40 + ac] = *(const uint4*)&gA[(size_t)ar * 1024 + k0 + ac];
#pragma unroll
        for (int h = 0; h < 4; ++h) {   // B: 1024 chunks of 16 B
            const int c  = tid + h * 256;
            const int br = c >> 2;
            const int bc = (c & 3) * 8;
            *(uint4*)&sB[br * 40 + bc] = *(const uint4*)&gB[(size_t)br * 1024 + k0 + bc];
        }
#endif
        __syncthreads();

        // A fragment: lane holds row M=lr, K = kh*8+{0..7} and kh*8+{16..23}
        v16bf af[2];
#pragma unroll
        for (int i = 0; i < 2; ++i) {
            const unsigned short* p = &sA[(wr * 32 + i * 16 + lr) * 40 + kh * 8];
            Q32 q;
            q.a = *(const uint4*)p;
            q.b = *(const uint4*)(p + 16);
            af[i] = __builtin_bit_cast(v16bf, q);
        }
        // B fragment: lane holds col N=lr, K = kh*8+{0..7} and kh*8+{16..23}
        v16bf bfm[4];
#pragma unroll
        for (int j = 0; j < 4; ++j) {
            const unsigned short* p = &sB[(wc * 64 + j * 16 + lr) * 40 + kh * 8];
            Q32 q;
            q.a = *(const uint4*)p;
            q.b = *(const uint4*)(p + 16);
            bfm[j] = __builtin_bit_cast(v16bf, q);
        }
#pragma unroll
        for (int i = 0; i < 2; ++i)
#pragma unroll
            for (int j = 0; j < 4; ++j)
                acc[i][j] = __builtin_amdgcn_wmma_f32_16x16x32_bf16(
                    false, af[i], false, bfm[j], (short)0, acc[i][j], false, false);
        __syncthreads();
    }

    // Epilogue: iso = |scale| * sqrt(max(2-2*sim,0)) / sqrt(2)
    const float dsv = fabsf(dscale[0]);
    const float is2 = 0.70710678118654752f;
#pragma unroll
    for (int i = 0; i < 2; ++i) {
#pragma unroll
        for (int j = 0; j < 4; ++j) {
            const int rbase = M0 + wr * 32 + i * 16 + kh * 8;  // C/D layout: M = v + half*8
            const int col   = N0 + wc * 64 + j * 16 + lr;      //            N = lane%16
            if (col < 1000) {
#pragma unroll
                for (int v = 0; v < 8; ++v) {
                    const float s = acc[i][j][v];
                    const float d = fast_sqrtf(fmaxf(2.0f - 2.0f * s, 0.0f)) * is2;
                    out[(size_t)(rbase + v) * 1000 + col] = dsv * d;
                }
            }
        }
    }
}

// One block per row: logits = -(iso + mean_c(iso)) / temperature, in place.
__global__ __launch_bounds__(256) void finalize_kernel(
    float* __restrict__ out, const float* __restrict__ temp, int C) {
    __shared__ float red[256];
    const int t  = threadIdx.x;
    float* rowp  = out + (size_t)blockIdx.x * C;

    float vals[4];
    int   idx[4];
    int   n = 0;
    float local = 0.0f;
    for (int c = t; c < C; c += 256) {
        vals[n] = rowp[c];
        idx[n]  = c;
        local  += vals[n];
        ++n;
    }
    red[t] = local;
    __syncthreads();
    for (int off = 128; off; off >>= 1) {
        if (t < off) red[t] += red[t + off];
        __syncthreads();
    }
    const float mean = red[0] / (float)C;
    const float invT = 1.0f / temp[0];
    for (int i = 0; i < n; ++i) rowp[idx[i]] = -(vals[i] + mean) * invT;
}

extern "C" void kernel_launch(void* const* d_in, const int* in_sizes, int n_in,
                              void* d_out, int out_size, void* d_ws, size_t ws_size,
                              hipStream_t stream) {
    const float* features   = (const float*)d_in[0];  // [16384,1024]
    const float* prototypes = (const float*)d_in[1];  // [1000,1024]
    const float* dscale     = (const float*)d_in[2];  // [1]
    const float* temp       = (const float*)d_in[3];  // [1]
    float*       out        = (float*)d_out;          // [16384,1000]

    unsigned short* wsA = (unsigned short*)d_ws;            // bf16 features
    unsigned short* wsP = wsA + (size_t)16384 * 1024;       // bf16 prototypes (padded)

    normalize_rows_kernel<<<16384, 256, 0, stream>>>(features, wsA, 16384);
    normalize_rows_kernel<<<1024, 256, 0, stream>>>(prototypes, wsP, 1000);

    dim3 grid(4, 256);  // N tiles (1024/256) x M tiles (16384/64)
    gemm_iso_kernel<<<grid, 256, 0, stream>>>(wsA, wsP, dscale, out);

    finalize_kernel<<<16384, 256, 0, stream>>>(out, temp, 1000);
}
